// LSTMAttentionDot_60060822667550
// MI455X (gfx1250) — matrix operand order repaired and checked by reading, not compiled, not run on hardware
//
#include <hip/hip_runtime.h>
#include <math.h>

typedef __bf16 bf16;
typedef __attribute__((ext_vector_type(16))) __bf16 v16bf;
typedef __attribute__((ext_vector_type(8)))  __bf16 v8bf;
typedef __attribute__((ext_vector_type(8)))  float   v8f;

#define T_DIM 512
#define B_DIM 32
#define D_DIM 1024
#define H_DIM 1024
#define S_DIM 512

// ---------- helpers ----------

__device__ __forceinline__ bf16 f2bf(float f) {
  union { float f; unsigned u; } v; v.f = f;
  unsigned r = (v.u + 0x7fffu + ((v.u >> 16) & 1u)) >> 16;
  unsigned short s = (unsigned short)r;
  return __builtin_bit_cast(bf16, s);
}

__device__ __forceinline__ float sigmf(float x) { return 1.0f / (1.0f + __expf(-x)); }

// A-fragment, 16-bit 16x32 (ISA 7.12.2 table):
// lanes 0-15:  M=lane,    slots 0-7 -> K=k0+0..7,   slots 8-15 -> K=k0+16..23
// lanes 16-31: M=lane-16, slots 0-7 -> K=k0+8..15,  slots 8-15 -> K=k0+24..31
__device__ __forceinline__ v16bf load_fragA(const bf16* __restrict__ base, int ld,
                                            int row0, int k0, int lane) {
  int m  = lane & 15;
  int kb = (lane >> 4) << 3; // 0 or 8
  const bf16* p = base + (size_t)(row0 + m) * ld + k0 + kb;
  v8bf lo = *(const v8bf*)(p);        // K = k0+kb   .. +7
  v8bf hi = *(const v8bf*)(p + 16);   // K = k0+16+kb.. +7
  v16bf r;
#pragma unroll
  for (int i = 0; i < 8; ++i) { r[i] = lo[i]; r[8 + i] = hi[i]; }
  return r;
}

// B-fragment, 16-bit 32x16 ("one row striped across lanes within one VGPR"):
// lane = N (mod 16); lanes 0-15 hold K=k0+0..7 (slots 0-7), k0+16..23 (slots 8-15),
// lanes 16-31 hold K=k0+8..15, k0+24..31 (same order as A). With B[k,n] = W[n,k]
// (W row-major [N,K]) this is two b128 loads from one W row.
__device__ __forceinline__ v16bf load_fragB(const bf16* __restrict__ Wrow, int ld,
                                            int col0, int k0, int lane) {
  int n   = lane & 15;
  int kb2 = (lane >> 4) << 3; // 0 or 8
  const bf16* p = Wrow + (size_t)(col0 + n) * ld + k0 + kb2;
  v8bf lo = *(const v8bf*)(p);       // K = k0+kb2   .. +7
  v8bf hi = *(const v8bf*)(p + 16);  // K = k0+16+kb2.. +7
  v16bf r;
#pragma unroll
  for (int i = 0; i < 8; ++i) { r[i] = lo[i]; r[8 + i] = hi[i]; }
  return r;
}

// ---------- conversion kernels ----------

__global__ void f32_to_bf16(const float* __restrict__ src, bf16* __restrict__ dst, size_t n) {
  size_t i = (size_t)blockIdx.x * blockDim.x + threadIdx.x;
  size_t stride = (size_t)gridDim.x * blockDim.x;
  for (; i < n; i += stride) dst[i] = f2bf(src[i]);
}

__global__ void init_state(const float* __restrict__ h0, const float* __restrict__ c0,
                           bf16* __restrict__ hx_bf, float* __restrict__ cx, int n) {
  int i = blockIdx.x * blockDim.x + threadIdx.x;
  if (i < n) { hx_bf[i] = f2bf(h0[i]); cx[i] = c0[i]; }
}

// ---------- generic WMMA bf16 GEMM: C[m,n] = act( sum_k A[m,k]*W[n,k] + bias[n] + Cadd[m,n] )
// A: [M,K] bf16 row-major (lda=K); W: [N,K] bf16 row-major (ldb=K); C: [M,N] f32.
// One wave per 16x64 tile; grid = (N/64, M/16). A fragment reused across 4 WMMAs.
__global__ void wmma_gemm_bf16(const bf16* __restrict__ A, const bf16* __restrict__ Bw,
                               const float* __restrict__ bias, const float* __restrict__ Cadd,
                               float* __restrict__ C, bf16* __restrict__ Cbf,
                               int N, int K, int act) {
  int lane = threadIdx.x;
  int row0 = blockIdx.y << 4;
  int col0 = blockIdx.x << 6;

  v8f acc[4];
#pragma unroll
  for (int j = 0; j < 4; ++j)
#pragma unroll
    for (int e = 0; e < 8; ++e) acc[j][e] = 0.0f;

  for (int k0 = 0; k0 < K; k0 += 32) {
    v16bf a = load_fragA(A, K, row0, k0, lane);
#pragma unroll
    for (int j = 0; j < 4; ++j) {
      v16bf b = load_fragB(Bw, K, col0 + (j << 4), k0, lane);
      acc[j] = __builtin_amdgcn_wmma_f32_16x16x32_bf16(
          /*neg_a=*/false, a, /*neg_b=*/false, b,
          /*c_mod=*/(short)0, acc[j], /*reuse_a=*/false, /*reuse_b=*/false);
    }
  }

  // C/D layout: lane -> N=lane&15; VGPR v -> M = v (+8 for lanes 16-31)
  int n    = lane & 15;
  int rsel = (lane >> 4) << 3;
#pragma unroll
  for (int v = 0; v < 8; ++v) {
    int r = row0 + v + rsel;
#pragma unroll
    for (int j = 0; j < 4; ++j) {
      int c = col0 + (j << 4) + n;
      float val = acc[j][v];
      if (bias) val += bias[c];
      if (Cadd) val += Cadd[(size_t)r * N + c];
      if (act)  val = tanhf(val);
      C[(size_t)r * N + c] = val;
      if (Cbf) Cbf[(size_t)r * N + c] = f2bf(val);
    }
  }
}

// ---------- LSTM cell elementwise ----------

__global__ void lstm_cell(const float* __restrict__ gates, float* __restrict__ cx,
                          bf16* __restrict__ hy_bf, bf16* __restrict__ concat_bf) {
  int i = blockIdx.x * blockDim.x + threadIdx.x;  // over B*H
  int b = i >> 10;
  int h = i & (H_DIM - 1);
  const float* g = gates + (size_t)b * 4 * H_DIM;
  float ig = g[h];
  float fg = g[H_DIM + h];
  float gg = g[2 * H_DIM + h];
  float og = g[3 * H_DIM + h];
  float cy = sigmf(fg) * cx[i] + sigmf(ig) * tanhf(gg);
  cx[i] = cy;
  float hy = sigmf(og) * tanhf(cy);
  hy_bf[i] = f2bf(hy);
  concat_bf[(size_t)b * 2 * H_DIM + H_DIM + h] = f2bf(hy);  // concat = [wc, hy]
}

// ---------- SoftDot attention: scores -> softmax -> weighted context ----------
// One block (512 threads) per batch row b. ctx is [S,B,H] f32 (L2-resident, 64MB).

__global__ void attention(const float* __restrict__ ctx, const float* __restrict__ target,
                          bf16* __restrict__ concat_bf) {
  __shared__ float sc[S_DIM];
  __shared__ float red[512];
  int b = blockIdx.x;
  int tid = threadIdx.x;

  // scores[s] = ctx[s,b,:] . target[b,:]
  {
    const float4* crow = (const float4*)(ctx + ((size_t)tid * B_DIM + b) * H_DIM);
    const float4* trow = (const float4*)(target + (size_t)b * H_DIM);
    float acc = 0.0f;
    for (int k = 0; k < H_DIM / 4; ++k) {
      float4 cv = crow[k], tv = trow[k];
      acc += cv.x * tv.x + cv.y * tv.y + cv.z * tv.z + cv.w * tv.w;
    }
    sc[tid] = acc;
  }
  __syncthreads();

  // max-reduce
  red[tid] = sc[tid];
  __syncthreads();
  for (int off = 256; off > 0; off >>= 1) {
    if (tid < off) red[tid] = fmaxf(red[tid], red[tid + off]);
    __syncthreads();
  }
  float mx = red[0];
  __syncthreads();

  // exp + sum-reduce
  float e = __expf(sc[tid] - mx);
  red[tid] = e;
  __syncthreads();
  for (int off = 256; off > 0; off >>= 1) {
    if (tid < off) red[tid] += red[tid + off];
    __syncthreads();
  }
  float inv = 1.0f / red[0];
  __syncthreads();
  sc[tid] = e * inv;           // attn weights
  __syncthreads();

  // wc[b,h] = sum_s attn[s] * ctx[s,b,h]; 512 threads x 2 h each (coalesced over h)
  int h1 = tid, h2 = tid + 512;
  float a1 = 0.0f, a2 = 0.0f;
  for (int s = 0; s < S_DIM; ++s) {
    float w = sc[s];
    const float* cr = ctx + ((size_t)s * B_DIM + b) * H_DIM;
    a1 += w * cr[h1];
    a2 += w * cr[h2];
  }
  concat_bf[(size_t)b * 2 * H_DIM + h1] = f2bf(a1);
  concat_bf[(size_t)b * 2 * H_DIM + h2] = f2bf(a2);
}

// ---------- host launch ----------

extern "C" void kernel_launch(void* const* d_in, const int* in_sizes, int n_in,
                              void* d_out, int out_size, void* d_ws, size_t ws_size,
                              hipStream_t stream) {
  (void)in_sizes; (void)n_in; (void)out_size; (void)ws_size;
  const float* x     = (const float*)d_in[0];
  const float* h0    = (const float*)d_in[1];
  const float* c0    = (const float*)d_in[2];
  const float* ctx   = (const float*)d_in[3];
  const float* W_ih  = (const float*)d_in[4];
  const float* b_ih  = (const float*)d_in[5];
  const float* W_hh  = (const float*)d_in[6];
  const float* b_hh  = (const float*)d_in[7];
  const float* W_in  = (const float*)d_in[8];
  const float* W_out = (const float*)d_in[9];
  float* out = (float*)d_out;

  const int T = T_DIM, B = B_DIM, D = D_DIM, H = H_DIM;

  // workspace carve-out
  char* w = (char*)d_ws;
  size_t off = 0;
  auto carve = [&](size_t bytes) -> void* {
    void* p = w + off;
    off = (off + bytes + 255) & ~(size_t)255;
    return p;
  };
  bf16*  Wih_bf    = (bf16*)carve((size_t)4 * H * D * sizeof(bf16));
  bf16*  Whh_bf    = (bf16*)carve((size_t)4 * H * H * sizeof(bf16));
  bf16*  Win_bf    = (bf16*)carve((size_t)H * H * sizeof(bf16));
  bf16*  Wout_bf   = (bf16*)carve((size_t)H * 2 * H * sizeof(bf16));
  bf16*  x_bf      = (bf16*)carve((size_t)T * B * D * sizeof(bf16));
  float* pre       = (float*)carve((size_t)T * B * 4 * H * sizeof(float));
  bf16*  hx_bf     = (bf16*)carve((size_t)B * H * sizeof(bf16));
  bf16*  hy_bf     = (bf16*)carve((size_t)B * H * sizeof(bf16));
  float* cx        = (float*)carve((size_t)B * H * sizeof(float));
  float* gates     = (float*)carve((size_t)B * 4 * H * sizeof(float));
  float* target    = (float*)carve((size_t)B * H * sizeof(float));
  bf16*  concat_bf = (bf16*)carve((size_t)B * 2 * H * sizeof(bf16));

  // 1) f32 -> bf16 conversions
  f32_to_bf16<<<2048, 256, 0, stream>>>(W_ih,  Wih_bf,  (size_t)4 * H * D);
  f32_to_bf16<<<2048, 256, 0, stream>>>(W_hh,  Whh_bf,  (size_t)4 * H * H);
  f32_to_bf16<<<1024, 256, 0, stream>>>(W_in,  Win_bf,  (size_t)H * H);
  f32_to_bf16<<<1024, 256, 0, stream>>>(W_out, Wout_bf, (size_t)H * 2 * H);
  f32_to_bf16<<<4096, 256, 0, stream>>>(x,     x_bf,    (size_t)T * B * D);
  init_state<<<(B * H) / 256, 256, 0, stream>>>(h0, c0, hx_bf, cx, B * H);

  // 2) hoisted input GEMM: pre[t,b,:] = x[t,b,:] @ W_ih^T + b_ih   (M=16384, N=4096, K=1024)
  wmma_gemm_bf16<<<dim3((4 * H) / 64, (T * B) / 16), 32, 0, stream>>>(
      x_bf, Wih_bf, b_ih, nullptr, pre, nullptr, 4 * H, D, 0);

  // 3) recurrence
  for (int t = 0; t < T; ++t) {
    const float* pre_t = pre + (size_t)t * B * 4 * H;
    // gates = hx @ W_hh^T + b_hh + pre[t]   (M=32, N=4096, K=1024)
    wmma_gemm_bf16<<<dim3((4 * H) / 64, B / 16), 32, 0, stream>>>(
        hx_bf, Whh_bf, b_hh, pre_t, gates, nullptr, 4 * H, H, 0);
    // LSTM cell elementwise -> cx (in place), hy_bf, concat[:,H:2H]
    lstm_cell<<<(B * H) / 256, 256, 0, stream>>>(gates, cx, hy_bf, concat_bf);
    // target = hy @ W_in^T   (M=32, N=1024, K=1024)
    wmma_gemm_bf16<<<dim3(H / 64, B / 16), 32, 0, stream>>>(
        hy_bf, Win_bf, nullptr, nullptr, target, nullptr, H, H, 0);
    // softmax attention + weighted context -> concat[:,0:H]
    attention<<<B, 512, 0, stream>>>(ctx, target, concat_bf);
    // h_tilde = tanh(concat @ W_out^T) -> out[t] (f32) and hx_bf (next step)
    wmma_gemm_bf16<<<dim3(H / 64, B / 16), 32, 0, stream>>>(
        concat_bf, Wout_bf, nullptr, nullptr, out + (size_t)t * B * H, hx_bf, H, 2 * H, 1);
  }

  // 4) hT = out[T-1], cT = cx
  hipMemcpyAsync(out + (size_t)T * B * H, out + (size_t)(T - 1) * B * H,
                 (size_t)B * H * sizeof(float), hipMemcpyDeviceToDevice, stream);
  hipMemcpyAsync(out + (size_t)T * B * H + (size_t)B * H, cx,
                 (size_t)B * H * sizeof(float), hipMemcpyDeviceToDevice, stream);
}
